// Net_70153995813115
// MI455X (gfx1250) — matrix operand order repaired and checked by reading, not compiled, not run on hardware
//
#include <hip/hip_runtime.h>
#include <hip/hip_bf16.h>
#include <math.h>

typedef __attribute__((ext_vector_type(16))) _Float16 v16h;
typedef __attribute__((ext_vector_type(8)))  float    v8f;

#define NEG_SLOPE 0.01f

__device__ __forceinline__ float lrelu(float v) { return v > 0.f ? v : v * NEG_SLOPE; }

// Monotonic float<->uint mapping so unsigned atomicMax == float max.
__device__ __forceinline__ unsigned f2ord(float f) {
    unsigned u = __float_as_uint(f);
    return (u & 0x80000000u) ? ~u : (u | 0x80000000u);
}
__device__ __forceinline__ float ord2f(unsigned u) {
    return (u & 0x80000000u) ? __uint_as_float(u & 0x7FFFFFFFu) : __uint_as_float(~u);
}

union AV { v16h v; _Float16 h[16]; unsigned u[8]; };

// ---------------------------------------------------------------------------
// Pre-swizzle weights into this code's WMMA B lane layout (contiguous K per lane half):
//   tile (chunk c, ntile nt): 32 lanes x 8 dwords, dword r = f16 pair
//   (K = c*32 + 2r + (lane>=16 ? 16 : 0), K+1) at column n = nt*16 + lane%16.
// Each wave then fetches its B tile as one 32-byte contiguous per-lane read.
// w1: 3 layers x 3 chunks x 4 nt (71 K-rows zero-padded to 96); w2: 3 x 2 x 4.
// ---------------------------------------------------------------------------
__global__ void k_pack_b(const float* __restrict__ w1, const float* __restrict__ w2,
                         unsigned* __restrict__ w1b, unsigned* __restrict__ w2b) {
    int i = blockIdx.x * blockDim.x + threadIdx.x;
    const int n1 = 3 * 3 * 4 * 32 * 8;   // 9216 dwords
    const int n2 = 3 * 2 * 4 * 32 * 8;   // 6144 dwords
    if (i < n1) {
        int r = i & 7, lane = (i >> 3) & 31, nt = (i >> 8) & 3;
        int c = (i >> 10) % 3, l = i / 3072;
        int k = c * 32 + 2 * r + ((lane >= 16) ? 16 : 0);
        int n = nt * 16 + (lane & 15);
        float f0 = (k     < 71) ? w1[((size_t)l * 71 + k    ) * 64 + n] : 0.f;
        float f1 = (k + 1 < 71) ? w1[((size_t)l * 71 + k + 1) * 64 + n] : 0.f;
        union { unsigned u; _Float16 h[2]; } p;
        p.h[0] = (_Float16)f0; p.h[1] = (_Float16)f1;
        w1b[i] = p.u;
    } else if (i < n1 + n2) {
        int j = i - n1;
        int r = j & 7, lane = (j >> 3) & 31, nt = (j >> 8) & 3;
        int c = (j >> 10) % 2, l = j / 2048;
        int k = c * 32 + 2 * r + ((lane >= 16) ? 16 : 0);
        int n = nt * 16 + (lane & 15);
        union { unsigned u; _Float16 h[2]; } p;
        p.h[0] = (_Float16)w2[((size_t)l * 64 + k    ) * 64 + n];
        p.h[1] = (_Float16)w2[((size_t)l * 64 + k + 1) * 64 + n];
        w2b[j] = p.u;
    }
}

// h0 = emb[x]; also the first 64 columns of the concat buffer.
__global__ void k_gather(const int* __restrict__ x, const float* __restrict__ emb,
                         float* __restrict__ h, float* __restrict__ feats, int N) {
    int idx = blockIdx.x * blockDim.x + threadIdx.x;
    if (idx >= N * 64) return;
    int n = idx >> 6, f = idx & 63;
    float v = emb[(size_t)x[n] * 64 + f];
    h[idx] = v;
    feats[(size_t)n * 256 + f] = v;
}

// Per-layer init: hnext=0, hdotar[n] = <h[n], att_r>, amax floor, denom=0.
__global__ void k_layer_init(const float* __restrict__ h, const float* __restrict__ ar,
                             float* __restrict__ hdotar, unsigned* __restrict__ amax,
                             float* __restrict__ denom, float* __restrict__ hnext, int N) {
    int idx = blockIdx.x * blockDim.x + threadIdx.x;
    if (idx < N * 64) hnext[idx] = 0.f;
    if (idx < N) {
        const float* hp = h + (size_t)idx * 64;
        float s = 0.f;
        #pragma unroll 8
        for (int f = 0; f < 64; ++f) s += hp[f] * ar[f];
        hdotar[idx] = s;
        amax[idx]   = 0u;      // below ord(-inf): smaller than any real alpha
        denom[idx]  = 0.f;
    }
}

// ---------------------------------------------------------------------------
// GEMM1: one wave per 16-edge tile. xj = leaky_relu([h[src]|edge_attr] @ w1)
// via v_wmma_f32_16x16x32_f16 (3 K-chunks x 4 N-tiles). Epilogue stores xj as
// f16 row-major (dword-pair friendly for GEMM2) and reduces xj . att_l.
// ---------------------------------------------------------------------------
__global__ void k_edge_gemm1(const int* __restrict__ src, const float* __restrict__ h,
                             const float* __restrict__ eattr, const v16h* __restrict__ w1b,
                             const float* __restrict__ al, _Float16* __restrict__ xj,
                             float* __restrict__ apart, int ntiles, int E) {
    int lane = threadIdx.x & 31;
    int tile = blockIdx.x * (blockDim.x >> 5) + (threadIdx.x >> 5);
    if (tile >= ntiles) return;
    int m  = lane & 15;          // A row / D column within tile
    int hi = lane >> 4;          // lane half selects K sub-block
    int ebase = tile * 16;
    int e  = ebase + m; if (e >= E) e = E - 1;

    const float2* hp2 = (const float2*)(h + (size_t)src[e] * 64);
    const float*  ep  = eattr + (size_t)e * 7;

    v8f acc[4] = {};
    #pragma unroll
    for (int c = 0; c < 3; ++c) {
        // A-matrix 16x32 f16 layout: lane m, K pairs; lane half hi covers K 16*hi..16*hi+15.
        AV a;
        #pragma unroll
        for (int r = 0; r < 8; ++r) {
            int k0 = c * 32 + 2 * r + hi * 16;
            float f0, f1;
            if (c < 2) { float2 p = hp2[k0 >> 1]; f0 = p.x; f1 = p.y; }
            else {
                f0 = (k0     < 71) ? ep[k0 - 64] : 0.f;
                f1 = (k0 + 1 < 71) ? ep[k0 - 63] : 0.f;
            }
            a.h[2 * r]     = (_Float16)f0;
            a.h[2 * r + 1] = (_Float16)f1;
        }
        #pragma unroll
        for (int nt = 0; nt < 4; ++nt) {
            v16h b = w1b[(size_t)((c * 4 + nt) * 32 + lane)];   // pre-swizzled tile
            acc[nt] = __builtin_amdgcn_wmma_f32_16x16x32_f16(
                false, a.v, false, b, (short)0, acc[nt], false, false);
        }
    }

    // Epilogue: leaky_relu, store xj (f16, row-major per tile), alpha partials.
    float part[8];
    #pragma unroll
    for (int r = 0; r < 8; ++r) part[r] = 0.f;
    #pragma unroll
    for (int nt = 0; nt < 4; ++nt) {
        float alv = al[nt * 16 + m];
        #pragma unroll
        for (int r = 0; r < 8; ++r) {
            float v = lrelu(acc[nt][r]);
            int mrow = r + hi * 8;                    // D layout: VGPR r -> row
            if (ebase + mrow < E)
                xj[((size_t)(ebase + mrow)) * 64 + nt * 16 + m] = (_Float16)v;
            part[r] += v * alv;
        }
    }
    #pragma unroll
    for (int r = 0; r < 8; ++r) {
        float v = part[r];
        v += __shfl_xor(v, 1); v += __shfl_xor(v, 2);
        v += __shfl_xor(v, 4); v += __shfl_xor(v, 8);
        int er = ebase + r + hi * 8;
        if (m == 0 && er < E) apart[er] = v;
    }
}

// alpha = leaky_relu(apart + <h[dst],att_r>); segment max via ordered-uint atomicMax.
__global__ void k_edge_alpha(const float* __restrict__ apart, const float* __restrict__ hdotar,
                             const int* __restrict__ dst, float* __restrict__ alpha,
                             unsigned* __restrict__ amax, int E) {
    int e = blockIdx.x * blockDim.x + threadIdx.x;
    if (e >= E) return;
    int d = dst[e];
    float v = lrelu(apart[e] + hdotar[d]);
    alpha[e] = v;
    atomicMax(amax + d, f2ord(v));
}

// ea = exp(alpha - amax[dst]); segment sum via f32 atomicAdd (L2-resident).
__global__ void k_edge_exp(const float* __restrict__ alpha, const unsigned* __restrict__ amax,
                           const int* __restrict__ dst, float* __restrict__ ea,
                           float* __restrict__ denom, int E) {
    int e = blockIdx.x * blockDim.x + threadIdx.x;
    if (e >= E) return;
    int d = dst[e];
    float a = expf(alpha[e] - ord2f(amax[d]));
    ea[e] = a;
    atomicAdd(denom + d, a);
}

// ---------------------------------------------------------------------------
// GEMM2: y = (xj @ w2) * softmax-coef, scattered by atomicAdd into hnext[dst].
// xj f16 rows are contiguous: A register pairs are single dword loads.
// Per-row coefficients computed once per lane and broadcast via shfl.
// ---------------------------------------------------------------------------
__global__ void k_edge_gemm2(const int* __restrict__ dst, const _Float16* __restrict__ xj,
                             const v16h* __restrict__ w2b, const float* __restrict__ ea,
                             const float* __restrict__ denom, float* __restrict__ hnext,
                             int ntiles, int E) {
    int lane = threadIdx.x & 31;
    int tile = blockIdx.x * (blockDim.x >> 5) + (threadIdx.x >> 5);
    if (tile >= ntiles) return;
    int m  = lane & 15;
    int hi = lane >> 4;
    int ebase = tile * 16;
    int e  = ebase + m; if (e >= E) e = E - 1;
    const unsigned* xpu = (const unsigned*)(xj + (size_t)e * 64);

    // Lane m's softmax coefficient (row m), broadcast to rows via shfl below.
    int   dm = dst[e];
    float cm = ea[e] / (denom[dm] + 1e-16f);

    v8f acc[4] = {};
    #pragma unroll
    for (int c = 0; c < 2; ++c) {
        AV a;
        #pragma unroll
        for (int r = 0; r < 8; ++r) {
            int k = c * 32 + 2 * r + hi * 16;
            a.u[r] = xpu[k >> 1];                    // f16 pair in one dword
        }
        #pragma unroll
        for (int nt = 0; nt < 4; ++nt) {
            v16h b = w2b[(size_t)((c * 4 + nt) * 32 + lane)];   // pre-swizzled tile
            acc[nt] = __builtin_amdgcn_wmma_f32_16x16x32_f16(
                false, a.v, false, b, (short)0, acc[nt], false, false);
        }
    }

    #pragma unroll
    for (int r = 0; r < 8; ++r) {
        int mrow = r + hi * 8;
        float cr = __shfl(cm, mrow);                 // lanes 0..15 hold rows 0..15
        int   dr = __shfl(dm, mrow);
        if (ebase + mrow < E) {
            #pragma unroll
            for (int nt = 0; nt < 4; ++nt) {
                float v = acc[nt][r] * cr;
                atomicAdd(&hnext[(size_t)dr * 64 + nt * 16 + m], v);
            }
        }
    }
}

// h = relu(hnext + gbias); also write into concat slice 64*(l+1).
__global__ void k_node_finish(const float* __restrict__ hnext, const float* __restrict__ gb,
                              float* __restrict__ h, float* __restrict__ feats,
                              int layer, int N) {
    int idx = blockIdx.x * blockDim.x + threadIdx.x;
    if (idx >= N * 64) return;
    int n = idx >> 6, f = idx & 63;
    float v = hnext[idx] + gb[f];
    v = v > 0.f ? v : 0.f;
    h[idx] = v;
    feats[(size_t)n * 256 + 64 * (layer + 1) + f] = v;
}

// Head: per node, register-blocked 256 -> 20 -> 1 with sigmoid.
__global__ void k_fc(const float* __restrict__ feats, const float* __restrict__ w1,
                     const float* __restrict__ b1, const float* __restrict__ w2,
                     const float* __restrict__ b2, float* __restrict__ out, int N) {
    int n = blockIdx.x * blockDim.x + threadIdx.x;
    if (n >= N) return;
    float z[20];
    #pragma unroll
    for (int j = 0; j < 20; ++j) z[j] = b1[j];
    const float* f = feats + (size_t)n * 256;
    for (int k = 0; k < 256; ++k) {
        float fv = f[k];
        #pragma unroll
        for (int j = 0; j < 20; ++j) z[j] += fv * w1[k * 20 + j];
    }
    float o = b2[0];
    #pragma unroll
    for (int j = 0; j < 20; ++j) {
        float zz = z[j] > 0.f ? z[j] : 0.f;
        o += zz * w2[j];
    }
    out[n] = 1.f / (1.f + expf(-o));
}

extern "C" void kernel_launch(void* const* d_in, const int* in_sizes, int n_in,
                              void* d_out, int out_size, void* d_ws, size_t ws_size,
                              hipStream_t stream) {
    const int N = in_sizes[0];
    const int E = in_sizes[1] / 2;

    const int*   x      = (const int*)  d_in[0];
    const int*   src    = (const int*)  d_in[1];
    const int*   dst    = src + E;
    const float* eattr  = (const float*)d_in[2];
    const float* emb    = (const float*)d_in[3];
    const float* lin1_w = (const float*)d_in[4];
    const float* att_l  = (const float*)d_in[5];
    const float* att_r  = (const float*)d_in[6];
    const float* lin2_w = (const float*)d_in[7];
    const float* gbias  = (const float*)d_in[8];
    const float* fc1_w  = (const float*)d_in[9];
    const float* fc1_b  = (const float*)d_in[10];
    const float* fc2_w  = (const float*)d_in[11];
    const float* fc2_b  = (const float*)d_in[12];
    float* out = (float*)d_out;

    // Workspace carve-out (256B aligned): ~245 MB total.
    char* w = (char*)d_ws;
    size_t off = 0;
    auto take = [&](size_t bytes) -> void* {
        void* p = w + off;
        off += (bytes + 255) & ~(size_t)255;
        return p;
    };
    unsigned* w1b    = (unsigned*)take((size_t)3 * 3 * 4 * 32 * 8 * 4);  // 36 KB
    unsigned* w2b    = (unsigned*)take((size_t)3 * 2 * 4 * 32 * 8 * 4);  // 24 KB
    float*    h      = (float*)   take((size_t)N * 64 * 4);
    float*    hnext  = (float*)   take((size_t)N * 64 * 4);
    float*    feats  = (float*)   take((size_t)N * 256 * 4);
    float*    hdotar = (float*)   take((size_t)N * 4);
    unsigned* amax   = (unsigned*)take((size_t)N * 4);
    float*    denom  = (float*)   take((size_t)N * 4);
    float*    apart  = (float*)   take((size_t)E * 4);
    float*    alpha  = (float*)   take((size_t)E * 4);
    float*    ea     = (float*)   take((size_t)E * 4);
    _Float16* xj     = (_Float16*)take((size_t)E * 64 * sizeof(_Float16));
    (void)ws_size; (void)n_in; (void)out_size;

    const int ntiles = (E + 15) / 16;            // E = 1.25M -> exact tiles
    const int wpb = 8;                            // waves per block (256 thr)
    dim3 blk(256);
    int gPack = (3 * 3 * 4 * 32 * 8 + 3 * 2 * 4 * 32 * 8 + 255) / 256;
    int gNode = (N * 64 + 255) / 256;
    int gEdge = (E + 255) / 256;
    int gTile = (ntiles + wpb - 1) / wpb;
    int gHead = (N + 255) / 256;

    k_pack_b<<<gPack, blk, 0, stream>>>(lin1_w, lin2_w, w1b, w2b);
    k_gather<<<gNode, blk, 0, stream>>>(x, emb, h, feats, N);

    for (int l = 0; l < 3; ++l) {
        const v16h* w1bl = (const v16h*)w1b + (size_t)l * 3 * 4 * 32;
        const v16h* w2bl = (const v16h*)w2b + (size_t)l * 2 * 4 * 32;
        const float* al = att_l + (size_t)l * 64;
        const float* ar = att_r + (size_t)l * 64;
        const float* gb = gbias + (size_t)l * 64;

        k_layer_init<<<gNode, blk, 0, stream>>>(h, ar, hdotar, amax, denom, hnext, N);
        k_edge_gemm1<<<gTile, blk, 0, stream>>>(src, h, eattr, w1bl, al, xj, apart, ntiles, E);
        k_edge_alpha<<<gEdge, blk, 0, stream>>>(apart, hdotar, dst, alpha, amax, E);
        k_edge_exp<<<gEdge, blk, 0, stream>>>(alpha, amax, dst, ea, denom, E);
        k_edge_gemm2<<<gTile, blk, 0, stream>>>(dst, xj, w2bl, ea, denom, hnext, ntiles, E);
        k_node_finish<<<gNode, blk, 0, stream>>>(hnext, gb, h, feats, l, N);
    }

    k_fc<<<gHead, blk, 0, stream>>>(feats, fc1_w, fc1_b, fc2_w, fc2_b, out, N);
}
